// HierarchicalAttention_54829552500811
// MI455X (gfx1250) — compile-verified, matrix operand order checked by reading, not run on hardware
//
#include <hip/hip_runtime.h>
#include <hip/hip_bf16.h>

typedef __bf16 bf16_t;
typedef __attribute__((ext_vector_type(16))) __bf16 v16bf;
typedef __attribute__((ext_vector_type(8)))  __bf16 v8bf;
typedef __attribute__((ext_vector_type(8)))  float  v8f;
typedef __attribute__((ext_vector_type(4)))  unsigned int u32x4;
typedef __attribute__((ext_vector_type(8)))  int i32x8;
typedef __attribute__((ext_vector_type(4)))  int i32x4;

#define D_MODEL 1024
#define NSEQ    2048
#define BATCH   2
#define NHEADS  16
#define DH      64
#define XYROWS  4096   // 2N-1 = 4095, padded to 4096 (row 4095 never gathered)
#define NKEYS   27     // 16 window + 11 tree levels

#define KSTEP     32   // K elements staged per TDM transfer
#define LDSTRIDE  40   // 32 elems + 8 pad elems -> 80B LDS row stride (bank-spread)

// ---------------------------------------------------------------------------
// Pack kernels: f32 -> bf16
// ---------------------------------------------------------------------------
__global__ __launch_bounds__(256) void pack_xy_kernel(
    const float* __restrict__ x, const float* __restrict__ y,
    bf16_t* __restrict__ xy) {
  long i = (long)blockIdx.x * blockDim.x + threadIdx.x;  // over B*4096*1024
  const long total = (long)BATCH * XYROWS * D_MODEL;
  if (i >= total) return;
  int d   = (int)(i % D_MODEL);
  long r  = i / D_MODEL;
  int pos = (int)(r % XYROWS);
  int b   = (int)(r / XYROWS);
  float v = 0.0f;
  if (pos < NSEQ)            v = x[((long)b * NSEQ + pos) * D_MODEL + d];
  else if (pos < 2*NSEQ - 1) v = y[((long)b * (NSEQ - 1) + (pos - NSEQ)) * D_MODEL + d];
  xy[i] = (bf16_t)v;
}

__global__ __launch_bounds__(256) void pack_w_kernel(
    const float* __restrict__ w, bf16_t* __restrict__ o, int n) {
  int i = blockIdx.x * blockDim.x + threadIdx.x;
  if (i < n) o[i] = (bf16_t)w[i];
}

// ---------------------------------------------------------------------------
// TDM: issue a 2D tile DMA (global -> LDS) via Tensor Data Mover.
// Descriptor packing per CDNA5 ISA 8.3/8.4 (D# group0 / group1).
//   tile_k: row length in elements (bf16), tile_rows: number of rows,
//   stride_elems: global row stride. Padding: after every 16 DWORDs (64B row)
//   insert 4 DWORDs (16B) -> 80B LDS row pitch == LDSTRIDE elements.
// ---------------------------------------------------------------------------
static __device__ __forceinline__ void tdm_load_2d(
    unsigned lds_off, const bf16_t* gptr, int tile_k, int tile_rows,
    int stride_elems) {
  unsigned long long ga = (unsigned long long)(uintptr_t)gptr;
  u32x4 g0;
  g0.x = 1u;                                     // count=1 (valid), user mode
  g0.y = lds_off;                                // lds_addr (bytes)
  g0.z = (unsigned)(ga & 0xFFFFFFFFu);           // global_addr[31:0]
  g0.w = (unsigned)((ga >> 32) & 0x1FFFFFFu)     // global_addr[56:32]
       | (2u << 30);                             // type = 2 ("image")
  i32x8 g1;
  const unsigned pad_interval_code = 3u;         // 16 DWORDs (64B) before pad
  const unsigned pad_amount_code   = 3u;         // 4 DWORDs (16B) of pad
  g1[0] = (int)((1u << 16)                       // data_size = 1 -> 2 bytes
              | (1u << 20)                       // pad_enable
              | (pad_interval_code << 22)
              | (pad_amount_code << 25));        // workgroup_mask = 0
  g1[1] = (int)(((unsigned)tile_k & 0xFFFFu) << 16);          // tensor_dim0 lo
  g1[2] = (int)((((unsigned)tile_k >> 16) & 0xFFFFu)          // tensor_dim0 hi
              | (((unsigned)tile_rows & 0xFFFFu) << 16));     // tensor_dim1 lo
  g1[3] = (int)((((unsigned)tile_rows >> 16) & 0xFFFFu)       // tensor_dim1 hi
              | (((unsigned)tile_k & 0xFFFFu) << 16));        // tile_dim0
  g1[4] = (int)((unsigned)tile_rows & 0xFFFFu);               // tile_dim1 (dim2=0)
  g1[5] = (int)(unsigned)stride_elems;                        // tensor_dim0_stride lo32
  g1[6] = 0;                                                  // stride hi / dim1_stride
  g1[7] = 0;
  i32x4 z4 = {0, 0, 0, 0};                       // groups 2/3 unused (2D tile)
  i32x8 z8 = {0, 0, 0, 0, 0, 0, 0, 0};           // extra group (clang-23 form)
  __builtin_amdgcn_tensor_load_to_lds(g0, g1, z4, z4, z8, 0);
}

// ---------------------------------------------------------------------------
// WMMA bf16 GEMM:  C[m][n] = sum_k A[m][k] * Bw[n][k]   (i.e.  A @ Bw^T)
// A: M x 1024 bf16 row-major;  Bw: 1024 x 1024 bf16 row-major.
// Block tile 128(M) x 128(N), 8 waves (4 in M x 2 in N), wave tile 32x64.
// TDM double-buffered LDS staging: wave 0 DMAs the next A/B 128x32 tiles
// while all waves run WMMA on the current buffer.
// ---------------------------------------------------------------------------
static __device__ __forceinline__ v16bf load_frag(const bf16_t* p, int ko0, int ko1) {
  union { v16bf f; v8bf h[2]; } u;
  u.h[0] = *(const v8bf*)(p + ko0);   // VGPRs 0..3: 8 contiguous bf16
  u.h[1] = *(const v8bf*)(p + ko1);   // VGPRs 4..7: 8 contiguous bf16
  return u.f;
}

__global__ __launch_bounds__(256) void gemm_bf16_kernel(
    const bf16_t* __restrict__ A, const bf16_t* __restrict__ Bw,
    float* __restrict__ Cf, bf16_t* __restrict__ Cb,
    const float* __restrict__ bias, int M) {
  __shared__ alignas(16) bf16_t shA[2][128 * LDSTRIDE];
  __shared__ alignas(16) bf16_t shB[2][128 * LDSTRIDE];

  const int K  = D_MODEL;
  const int NN = D_MODEL;
  const int lane = threadIdx.x & 31;
  const int wave = threadIdx.x >> 5;
  const int wm = wave & 3;             // 0..3  -> 32-row slabs
  const int wn = wave >> 2;            // 0..1  -> 64-col slabs
  const int m0 = blockIdx.y * 128 + wm * 32;
  const int n0 = blockIdx.x * 128 + wn * 64;
  const int r  = lane & 15;            // row within 16x16 tile
  const int h  = lane >> 4;            // half-wave selector

  const bf16_t* gA = A  + (long)(blockIdx.y * 128) * K;
  const bf16_t* gB = Bw + (long)(blockIdx.x * 128) * K;

  v8f acc[2][4];
  #pragma unroll
  for (int mt = 0; mt < 2; ++mt)
    #pragma unroll
    for (int nt = 0; nt < 4; ++nt)
      acc[mt][nt] = (v8f){0.f,0.f,0.f,0.f,0.f,0.f,0.f,0.f};

  const bool issuer = (wave == 0);
  if (issuer) {  // prologue: stage buffer 0
    tdm_load_2d((unsigned)(uintptr_t)&shA[0][0], gA, KSTEP, 128, K);
    tdm_load_2d((unsigned)(uintptr_t)&shB[0][0], gB, KSTEP, 128, K);
  }

  const int niter = K / KSTEP;
  for (int it = 0; it < niter; ++it) {
    const int cur = it & 1;
    if (issuer) {
      if (it + 1 < niter) {   // stage next buffer, then wait for current
        const int kb2 = (it + 1) * KSTEP;
        tdm_load_2d((unsigned)(uintptr_t)&shA[1 - cur][0], gA + kb2, KSTEP, 128, K);
        tdm_load_2d((unsigned)(uintptr_t)&shB[1 - cur][0], gB + kb2, KSTEP, 128, K);
        __builtin_amdgcn_s_wait_tensorcnt((short)2);
      } else {
        __builtin_amdgcn_s_wait_tensorcnt((short)0);
      }
    }
    __syncthreads();          // current buffer visible to all waves

    const bf16_t* tA = &shA[cur][0];
    const bf16_t* tB = &shB[cur][0];
    const int ko0 = h * 8;
    const int ko1 = 16 + h * 8;

    v16bf afrag[2], bfrag[4];
    #pragma unroll
    for (int mt = 0; mt < 2; ++mt)
      afrag[mt] = load_frag(tA + (wm * 32 + mt * 16 + r) * LDSTRIDE, ko0, ko1);
    #pragma unroll
    for (int nt = 0; nt < 4; ++nt)
      bfrag[nt] = load_frag(tB + (wn * 64 + nt * 16 + r) * LDSTRIDE, ko0, ko1);

    #pragma unroll
    for (int mt = 0; mt < 2; ++mt)
      #pragma unroll
      for (int nt = 0; nt < 4; ++nt)
        acc[mt][nt] = __builtin_amdgcn_wmma_f32_16x16x32_bf16(
            false, afrag[mt], false, bfrag[nt], (short)0, acc[mt][nt],
            false, false);

    __syncthreads();          // reads done before buffer is overwritten
  }

  // C/D layout: element c of v8f -> row = base_m + h*8 + c, col = base_n + r
  #pragma unroll
  for (int mt = 0; mt < 2; ++mt) {
    #pragma unroll
    for (int nt = 0; nt < 4; ++nt) {
      const int col = n0 + nt * 16 + r;
      const float bv = bias ? bias[col] : 0.0f;
      #pragma unroll
      for (int c = 0; c < 8; ++c) {
        const int row = m0 + mt * 16 + h * 8 + c;
        const long idx = (long)row * NN + col;
        if (Cf) Cf[idx] = acc[mt][nt][c] + bv;
        else    Cb[idx] = (bf16_t)acc[mt][nt][c];
      }
    }
  }
  (void)M;
}

// ---------------------------------------------------------------------------
// Gathered hierarchical attention: one wave per (b, n, h).
// Lane owns dims {lane, lane+32} of the 64-dim head.
// ---------------------------------------------------------------------------
__global__ __launch_bounds__(256) void attn_kernel(
    const bf16_t* __restrict__ Qb, const bf16_t* __restrict__ Kb,
    const bf16_t* __restrict__ Vb, bf16_t* __restrict__ Ob) {
  const int lane = threadIdx.x & 31;
  const int wid  = blockIdx.x * (blockDim.x >> 5) + (threadIdx.x >> 5);
  if (wid >= BATCH * NSEQ * NHEADS) return;
  const int hh = wid & (NHEADS - 1);
  const int n  = (wid >> 4) & (NSEQ - 1);
  const int b  = wid >> 15;

  const long qrow = ((long)b * NSEQ + n) * D_MODEL + hh * DH;
  const float q0 = (float)Qb[qrow + lane];
  const float q1 = (float)Qb[qrow + lane + 32];

  float s[NKEYS];
  int   idxs[NKEYS];

  #pragma unroll
  for (int t = 0; t < NKEYS; ++t) {
    int j; bool valid = true;
    if (t < 16) {
      j = n - 15 + t;                       // causal window i-15..i
      valid = (j >= 0);
      if (!valid) j = 0;
    } else {
      const int l = t - 16;                 // ancestor at level l
      j = 4096 - (2048 >> l) + (n >> (l + 1));
    }
    idxs[t] = j;
    const long krow = ((long)b * XYROWS + j) * D_MODEL + hh * DH;
    float part = q0 * (float)Kb[krow + lane] + q1 * (float)Kb[krow + lane + 32];
    #pragma unroll
    for (int off = 16; off > 0; off >>= 1)
      part += __shfl_xor(part, off, 32);
    s[t] = valid ? part * 0.125f : -3.0e38f;   // scale = 1/sqrt(64)
  }

  float m = s[0];
  #pragma unroll
  for (int t = 1; t < NKEYS; ++t) m = fmaxf(m, s[t]);
  float sum = 0.0f;
  #pragma unroll
  for (int t = 0; t < NKEYS; ++t) { s[t] = __expf(s[t] - m); sum += s[t]; }
  const float inv = 1.0f / sum;

  float o0 = 0.0f, o1 = 0.0f;
  #pragma unroll
  for (int t = 0; t < NKEYS; ++t) {
    const long vrow = ((long)b * XYROWS + idxs[t]) * D_MODEL + hh * DH;
    const float p = s[t];
    o0 += p * (float)Vb[vrow + lane];
    o1 += p * (float)Vb[vrow + lane + 32];
  }
  Ob[qrow + lane]      = (bf16_t)(o0 * inv);
  Ob[qrow + lane + 32] = (bf16_t)(o1 * inv);
}

// ---------------------------------------------------------------------------
// Launcher
// ---------------------------------------------------------------------------
extern "C" void kernel_launch(void* const* d_in, const int* in_sizes, int n_in,
                              void* d_out, int out_size, void* d_ws, size_t ws_size,
                              hipStream_t stream) {
  const float* x  = (const float*)d_in[0];
  const float* y  = (const float*)d_in[1];
  const float* Wq = (const float*)d_in[2];
  const float* Wk = (const float*)d_in[3];
  const float* Wv = (const float*)d_in[4];
  const float* Wo = (const float*)d_in[5];
  const float* bo = (const float*)d_in[6];
  float* out = (float*)d_out;

  char* ws = (char*)d_ws;
  size_t off = 0;
  auto alloc = [&](size_t bytes) -> void* {
    void* p = ws + off;
    off += (bytes + 255) & ~(size_t)255;
    return p;
  };
  const size_t DD = (size_t)D_MODEL * D_MODEL;
  bf16_t* XYb = (bf16_t*)alloc((size_t)BATCH * XYROWS * D_MODEL * 2);
  bf16_t* Wqb = (bf16_t*)alloc(DD * 2);
  bf16_t* Wkb = (bf16_t*)alloc(DD * 2);
  bf16_t* Wvb = (bf16_t*)alloc(DD * 2);
  bf16_t* Wob = (bf16_t*)alloc(DD * 2);
  bf16_t* Qb  = (bf16_t*)alloc((size_t)BATCH * NSEQ * D_MODEL * 2);
  bf16_t* Kb  = (bf16_t*)alloc((size_t)BATCH * XYROWS * D_MODEL * 2);
  bf16_t* Vb  = (bf16_t*)alloc((size_t)BATCH * XYROWS * D_MODEL * 2);
  bf16_t* Ob  = (bf16_t*)alloc((size_t)BATCH * NSEQ * D_MODEL * 2);

  // 1) pack inputs + weights to bf16
  {
    const long total = (long)BATCH * XYROWS * D_MODEL;
    pack_xy_kernel<<<(unsigned)((total + 255) / 256), 256, 0, stream>>>(x, y, XYb);
    const int nw = (int)DD;
    pack_w_kernel<<<(nw + 255) / 256, 256, 0, stream>>>(Wq, Wqb, nw);
    pack_w_kernel<<<(nw + 255) / 256, 256, 0, stream>>>(Wk, Wkb, nw);
    pack_w_kernel<<<(nw + 255) / 256, 256, 0, stream>>>(Wv, Wvb, nw);
    pack_w_kernel<<<(nw + 255) / 256, 256, 0, stream>>>(Wo, Wob, nw);
  }

  // 2) projections (WMMA bf16 GEMMs with TDM-staged LDS tiles)
  for (int b = 0; b < BATCH; ++b) {  // Q per batch (x rows at XYb[b*4096 + 0..2047])
    dim3 g(D_MODEL / 128, NSEQ / 128);
    gemm_bf16_kernel<<<g, 256, 0, stream>>>(
        XYb + (size_t)b * XYROWS * D_MODEL, Wqb,
        nullptr, Qb + (size_t)b * NSEQ * D_MODEL, nullptr, NSEQ);
  }
  {
    dim3 g(D_MODEL / 128, (BATCH * XYROWS) / 128);
    gemm_bf16_kernel<<<g, 256, 0, stream>>>(XYb, Wkb, nullptr, Kb, nullptr,
                                            BATCH * XYROWS);
    gemm_bf16_kernel<<<g, 256, 0, stream>>>(XYb, Wvb, nullptr, Vb, nullptr,
                                            BATCH * XYROWS);
  }

  // 3) gathered attention (VALU; <1% of FLOPs)
  {
    const int waves = BATCH * NSEQ * NHEADS;       // 65536
    attn_kernel<<<waves / 8, 256, 0, stream>>>(Qb, Kb, Vb, Ob);
  }

  // 4) output projection + bias -> f32 d_out
  {
    dim3 g(D_MODEL / 128, (BATCH * NSEQ) / 128);
    gemm_bf16_kernel<<<g, 256, 0, stream>>>(Ob, Wob, out, nullptr, bo,
                                            BATCH * NSEQ);
  }
  (void)in_sizes; (void)n_in; (void)out_size; (void)ws_size;
}
